// fusion_module_43611097923620
// MI455X (gfx1250) — matrix operand order repaired and checked by reading, not compiled, not run on hardware
//
#include <hip/hip_runtime.h>
#include <hip/hip_bf16.h>

// ---------------------------------------------------------------------------
// MaxSigmoidAttnBlock fusion for MI455X (gfx1250, wave32, WMMA).
//
// Shapes: B=16, CIN=EC=COUT=256, T=4096, N=64, GC=224, H=8, HC=32, K=3.
// Conv dominates (25.8 GFLOP) -> bf16x3-split WMMA (f32 accumulate), with the
// k=3 conv decomposed into 3 tap-GEMMs so all LDS->VGPR fragment builds are
// pure ds_load_b128 (bf16 K-pairs pre-packed, zero VALU unpack in hot loops).
// g (guide features) is pre-packed once into B-fragment layout so the 32K
// attention waves load fragments with global_load_b128 instead of converting.
// ---------------------------------------------------------------------------

typedef float   v8f   __attribute__((ext_vector_type(8)));
typedef float   v2f   __attribute__((ext_vector_type(2)));
typedef __bf16  v16bf __attribute__((ext_vector_type(16)));

// Split f32 into bf16 hi/lo (round-to-nearest-even), packed (lo<<16)|hi.
static __device__ __forceinline__ unsigned bf16split(float f) {
  unsigned u  = __float_as_uint(f);
  unsigned hi = (u + 0x7FFFu + ((u >> 16) & 1u)) & 0xFFFF0000u;
  float    fh = __uint_as_float(hi);
  float    r  = f - fh;
  unsigned v  = __float_as_uint(r);
  unsigned lo = (v + 0x7FFFu + ((v >> 16) & 1u)) >> 16;
  return (lo << 16) | (hi >> 16);
}
// Pack two splits into fragment words: element 2p -> [15:0], 2p+1 -> [31:16].
static __device__ __forceinline__ unsigned pack_hi(unsigned s0, unsigned s1) {
  return (s0 & 0xFFFFu) | (s1 << 16);
}
static __device__ __forceinline__ unsigned pack_lo(unsigned s0, unsigned s1) {
  return (s0 >> 16) | (s1 & 0xFFFF0000u);
}

// Fragment <-> raw-word views (frag VGPR p holds K-pair words).
union FragU {
  uint4    q[2];
  unsigned u[8];
  v16bf    v;
};
static __device__ __forceinline__ v16bf ldsfrag(const unsigned* p, int w0,
                                                int w1) {
  FragU f;
  f.q[0] = *(const uint4*)(p + w0);
  f.q[1] = *(const uint4*)(p + w1);
  return f.v;
}

// ---------------------------------------------------------------------------
// Kernel 1: guide_fc  g[b,n,e] = sum_k guide[b,n,k]*gw[e,k] + gb[e]
// One wave per 16x16 tile, V_WMMA_F32_16X16X4_F32, K=224 (56 steps).
// ---------------------------------------------------------------------------
__global__ void guide_fc_kernel(const float* __restrict__ guide,
                                const float* __restrict__ gw,
                                const float* __restrict__ gb,
                                float* __restrict__ g) {
  const int e0 = blockIdx.x * 16;   // EC tile (16 of 256)
  const int n0 = blockIdx.y * 16;   // N tile  (4 of 64)
  const int b  = blockIdx.z;
  const int l    = threadIdx.x;
  const int m    = l & 15;
  const int half = l >> 4;

  const float* arow = guide + (b * 64 + n0 + m) * 224;  // A[n,k]
  const float* brow = gw    + (e0 + m) * 224;           // B[k,e] = gw[e,k]

  v8f acc = {};
  for (int k0 = 0; k0 < 224; k0 += 4) {
    // f32 A 16x4: VGPR r holds K = r + 2*half;  B 4x16 mirrors it.
    v2f a, bv;
    a.x  = arow[k0 + 2 * half];
    a.y  = arow[k0 + 2 * half + 1];
    bv.x = brow[k0 + 2 * half];
    bv.y = brow[k0 + 2 * half + 1];
    acc = __builtin_amdgcn_wmma_f32_16x16x4_f32(false, a, false, bv,
                                                (short)0, acc, false, false);
  }

  const float bias = gb[e0 + m];
#pragma unroll
  for (int j = 0; j < 8; ++j) {
    const int row = n0 + j + 8 * half;             // C/D: M = vgpr + 8*half
    g[(b * 64 + row) * 256 + e0 + m] = acc[j] + bias;
  }
}

// ---------------------------------------------------------------------------
// Kernel 1b: pack g into attention B-fragment layout.
// Word w = [b][n][ec-pair cpx] = {bf16 g[...,2cpx+1] | bf16 g[...,2cpx]}.
// ---------------------------------------------------------------------------
__global__ void pack_g_kernel(const float* __restrict__ g,
                              unsigned* __restrict__ gpk_hi,
                              unsigned* __restrict__ gpk_lo) {
  const int idx = blockIdx.x * 256 + threadIdx.x;   // 131072 words
  const unsigned s0 = bf16split(g[2 * idx]);
  const unsigned s1 = bf16split(g[2 * idx + 1]);
  gpk_hi[idx] = pack_hi(s0, s1);
  gpk_lo[idx] = pack_lo(s0, s1);
}

// ---------------------------------------------------------------------------
// Kernel 2: attn[b,h,t] = sigmoid(max_n(sum_c x[b,h*32+c,t]*g[b,n,h*32+c])
//                                 / sqrt(32) + bias[h])
// 4 waves/block, one (b,h,16-t strip) per wave. bf16x3, K=HC=32 = one K-step.
// B frags: 2x global_load_b128 from pre-packed g.  A frags: pair-packed gather.
// ---------------------------------------------------------------------------
__global__ __launch_bounds__(128) void attn_kernel(
    const float* __restrict__ x, const unsigned* __restrict__ gpk_hi,
    const unsigned* __restrict__ gpk_lo, const float* __restrict__ attn_bias,
    float* __restrict__ attn) {
  const int wave = threadIdx.x >> 5;
  const int t0   = (blockIdx.x * 4 + wave) * 16;
  const int h    = blockIdx.y;
  const int b    = blockIdx.z;
  const int l    = threadIdx.x & 31;
  const int m    = l & 15;
  const int half = l >> 4;

  // A frag: A[t,c] = x[b, h*32+c, t0+t]; VGPR p holds K-pair (c0, c0+1).
  FragU ahi, alo;
#pragma unroll
  for (int p = 0; p < 8; ++p) {
    const int c0 = 16 * half + 2 * p;
    const size_t addr = ((size_t)b * 256 + h * 32 + c0) * 4096 + t0 + m;
    const unsigned s0 = bf16split(x[addr]);
    const unsigned s1 = bf16split(x[addr + 4096]);
    ahi.u[p] = pack_hi(s0, s1);
    alo.u[p] = pack_lo(s0, s1);
  }

  v8f vmax;
#pragma unroll
  for (int j = 0; j < 8; ++j) vmax[j] = -3.0e38f;

#pragma unroll
  for (int nt = 0; nt < 4; ++nt) {
    // B frag: word base = [b][n = nt*16 + m][ec-pair h*16 + 8*half + p]
    const int wbase = ((b * 64 + nt * 16 + m) * 128) + h * 16 + 8 * half;
    const v16bf bhi = ldsfrag(gpk_hi, wbase, wbase + 4);
    const v16bf blo = ldsfrag(gpk_lo, wbase, wbase + 4);
    v8f acc = {};
    acc = __builtin_amdgcn_wmma_f32_16x16x32_bf16(false, ahi.v, false, bhi,
                                                  (short)0, acc, false, false);
    acc = __builtin_amdgcn_wmma_f32_16x16x32_bf16(false, ahi.v, false, blo,
                                                  (short)0, acc, false, false);
    acc = __builtin_amdgcn_wmma_f32_16x16x32_bf16(false, alo.v, false, bhi,
                                                  (short)0, acc, false, false);
#pragma unroll
    for (int j = 0; j < 8; ++j) vmax[j] = fmaxf(vmax[j], acc[j]);
  }

  // Row-max over N: lanes within each 16-lane half hold distinct n, same row.
#pragma unroll
  for (int off = 8; off >= 1; off >>= 1) {
#pragma unroll
    for (int j = 0; j < 8; ++j)
      vmax[j] = fmaxf(vmax[j], __shfl_xor(vmax[j], off, 32));
  }

  const float bias = attn_bias[h];
  if (m == 0) {                          // lane 0 -> rows 0..7, lane 16 -> 8..15
#pragma unroll
    for (int j = 0; j < 8; ++j) {
      const float z = vmax[j] * 0.17677669529663687f + bias;  // 1/sqrt(32)
      const float s = 1.0f / (1.0f + __expf(-z));
      attn[((size_t)b * 8 + h) * 4096 + t0 + 8 * half + j] = s;
    }
  }
}

// ---------------------------------------------------------------------------
// Kernel 3: conv1d(k=3,pad=1) fused with bias + mask + attn scaling.
// Decomposed over the 3 taps:  out[co,t] = sum_koff  W_koff · x[:, t+koff-1].
// Each tap is a GEMM with K=CIN=256 (8 WMMA K-steps), bf16x3 split.
// All fragments are 2x ds_load_b128 from K-pair-packed LDS tiles; stride 132
// words (132 % 64 == 4) de-conflicts the 16 per-lane rows.
// WG = 256 threads (8 waves) -> 32co x 128t macro-tile; wave (ct,q) computes
// 16co x 32t (two 16x16 D tiles) accumulated across all 3 taps.
// ---------------------------------------------------------------------------
#define XSTR 132
__global__ __launch_bounds__(256) void conv_attn_kernel(
    const float* __restrict__ x, const unsigned char* __restrict__ mask,
    const float* __restrict__ pw, const float* __restrict__ pb,
    const float* __restrict__ attn, float* __restrict__ out) {
  const int tchunk = blockIdx.x * 128;
  const int co0    = blockIdx.y * 32;
  const int h      = blockIdx.y;           // COUT//H == 32 -> head = co/32
  const int b      = blockIdx.z;
  const int tid    = threadIdx.x;

  // Weights: [koff 0..2][co 0..31][ci-pair 0..127], bf16 pairs over ci.
  __shared__ __align__(16) unsigned wtile_hi[3 * 32 * XSTR];  // 50.7 KB
  __shared__ __align__(16) unsigned wtile_lo[3 * 32 * XSTR];  // 50.7 KB
  // x: [tt 0..129 (halo t0-1..t0+128)][ci-pair 0..127], bf16 pairs over ci.
  __shared__ __align__(16) unsigned xtile_hi[130 * XSTR];     // 68.6 KB
  __shared__ __align__(16) unsigned xtile_lo[130 * XSTR];     // 68.6 KB

  // ---- stage weights: word(koff,co,cp) = {W[co,2cp+1,koff] | W[co,2cp,koff]}
  for (int idx = tid; idx < 3 * 32 * 128; idx += 256) {
    const int cp   = idx & 127;
    const int co   = (idx >> 7) & 31;
    const int koff = idx >> 12;
    const size_t base = ((size_t)(co0 + co) * 256 + 2 * cp) * 3 + koff;
    const unsigned s0 = bf16split(pw[base]);
    const unsigned s1 = bf16split(pw[base + 3]);
    const int w = (koff * 32 + co) * XSTR + cp;
    wtile_hi[w] = pack_hi(s0, s1);
    wtile_lo[w] = pack_lo(s0, s1);
  }
  // ---- stage x interior rows tt=1..128 (gt always in-bounds, no div/branch)
#pragma unroll 4
  for (int w = 0; w < 64; ++w) {
    const int idx = w * 256 + tid;
    const int tt  = (idx & 127) + 1;
    const int cp  = idx >> 7;
    const int gt  = tchunk + (idx & 127);
    const unsigned s0 = bf16split(x[((size_t)b * 256 + 2 * cp) * 4096 + gt]);
    const unsigned s1 = bf16split(x[((size_t)b * 256 + 2 * cp + 1) * 4096 + gt]);
    xtile_hi[tt * XSTR + cp] = pack_hi(s0, s1);
    xtile_lo[tt * XSTR + cp] = pack_lo(s0, s1);
  }
  // ---- stage x halo rows tt=0 and tt=129
  {
    const int tt = (tid >> 7) * 129;               // 0 or 129
    const int cp = tid & 127;
    const int gt = tchunk - 1 + tt;
    const bool ok = (gt >= 0) & (gt < 4096);
    const float v0 = ok ? x[((size_t)b * 256 + 2 * cp) * 4096 + gt] : 0.0f;
    const float v1 = ok ? x[((size_t)b * 256 + 2 * cp + 1) * 4096 + gt] : 0.0f;
    const unsigned s0 = bf16split(v0);
    const unsigned s1 = bf16split(v1);
    xtile_hi[tt * XSTR + cp] = pack_hi(s0, s1);
    xtile_lo[tt * XSTR + cp] = pack_lo(s0, s1);
  }
  __syncthreads();

  const int l    = tid & 31;
  const int m    = l & 15;
  const int half = l >> 4;
  const int wave = tid >> 5;
  const int ct   = wave >> 2;              // co tile within WG (0..1)
  const int q    = wave & 3;               // 32-wide t quarter (0..3)

  v8f acc0 = {}, acc1 = {};
#pragma unroll
  for (int koff = 0; koff < 3; ++koff) {
    const int arow  = (koff * 32 + ct * 16 + m) * XSTR;     // A: co row
    const int brow0 = (q * 32 + m + koff) * XSTR;           // B: t row (halo)
    const int brow1 = brow0 + 16 * XSTR;
#pragma unroll
    for (int s = 0; s < 8; ++s) {                           // K = 256 = 8 x 32
      const int aq0 = arow + s * 16 + 8 * half;             // A quads
      const int aq1 = aq0 + 4;
      const int bq0 = s * 16 + 8 * half;                    // B quads (rel)
      const v16bf ahi = ldsfrag(wtile_hi, aq0, aq1);
      const v16bf alo = ldsfrag(wtile_lo, aq0, aq1);
      const v16bf b0h = ldsfrag(xtile_hi, brow0 + bq0, brow0 + bq0 + 4);
      const v16bf b0l = ldsfrag(xtile_lo, brow0 + bq0, brow0 + bq0 + 4);
      const v16bf b1h = ldsfrag(xtile_hi, brow1 + bq0, brow1 + bq0 + 4);
      const v16bf b1l = ldsfrag(xtile_lo, brow1 + bq0, brow1 + bq0 + 4);
      acc0 = __builtin_amdgcn_wmma_f32_16x16x32_bf16(false, ahi, false, b0h,
                                                     (short)0, acc0, false, false);
      acc0 = __builtin_amdgcn_wmma_f32_16x16x32_bf16(false, ahi, false, b0l,
                                                     (short)0, acc0, false, false);
      acc0 = __builtin_amdgcn_wmma_f32_16x16x32_bf16(false, alo, false, b0h,
                                                     (short)0, acc0, false, false);
      acc1 = __builtin_amdgcn_wmma_f32_16x16x32_bf16(false, ahi, false, b1h,
                                                     (short)0, acc1, false, false);
      acc1 = __builtin_amdgcn_wmma_f32_16x16x32_bf16(false, ahi, false, b1l,
                                                     (short)0, acc1, false, false);
      acc1 = __builtin_amdgcn_wmma_f32_16x16x32_bf16(false, alo, false, b1h,
                                                     (short)0, acc1, false, false);
    }
  }

  // Epilogue: (acc + pb) * mask * attn  -> out  (cols = t -> coalesced lanes)
  const int tA = tchunk + q * 32 + m;
  const int tB = tA + 16;
  const float attA = attn[((size_t)b * 8 + h) * 4096 + tA];
  const float attB = attn[((size_t)b * 8 + h) * 4096 + tB];
  const float mA = mask[(size_t)b * 4096 + tA] ? 1.0f : 0.0f;
  const float mB = mask[(size_t)b * 4096 + tB] ? 1.0f : 0.0f;
#pragma unroll
  for (int j = 0; j < 8; ++j) {
    const int co = co0 + ct * 16 + j + 8 * half;           // C/D: M = vgpr+8*half
    const float pbv = pb[co];
    out[((size_t)b * 256 + co) * 4096 + tA] = (acc0[j] + pbv) * mA * attA;
    out[((size_t)b * 256 + co) * 4096 + tB] = (acc1[j] + pbv) * mB * attB;
  }
}

// ---------------------------------------------------------------------------
extern "C" void kernel_launch(void* const* d_in, const int* in_sizes, int n_in,
                              void* d_out, int out_size, void* d_ws,
                              size_t ws_size, hipStream_t stream) {
  (void)in_sizes; (void)n_in; (void)out_size; (void)ws_size;
  const float*         x         = (const float*)d_in[0];
  const float*         guide     = (const float*)d_in[1];
  const unsigned char* mask      = (const unsigned char*)d_in[2];  // bool[B,1,T]
  const float*         gw        = (const float*)d_in[3];
  const float*         gb        = (const float*)d_in[4];
  const float*         attn_bias = (const float*)d_in[5];
  const float*         pw        = (const float*)d_in[6];
  const float*         pb        = (const float*)d_in[7];
  float*               out       = (float*)d_out;

  // Workspace layout (4 MB): g f32 | g packed hi | g packed lo | attn
  float*    gbuf    = (float*)d_ws;                  // 262144 floats
  unsigned* gpk_hi  = (unsigned*)(gbuf + 262144);    // 131072 words
  unsigned* gpk_lo  = gpk_hi + 131072;               // 131072 words
  float*    attnbuf = (float*)(gpk_lo + 131072);     // 524288 floats

  guide_fc_kernel<<<dim3(16, 4, 16), 32, 0, stream>>>(guide, gw, gb, gbuf);
  pack_g_kernel<<<512, 256, 0, stream>>>(gbuf, gpk_hi, gpk_lo);
  attn_kernel<<<dim3(64, 8, 16), 128, 0, stream>>>(x, gpk_hi, gpk_lo,
                                                   attn_bias, attnbuf);
  conv_attn_kernel<<<dim3(32, 8, 16), 256, 0, stream>>>(x, mask, pw, pb,
                                                        attnbuf, out);
}
